// TemporalDeformableBlock_75952201662777
// MI455X (gfx1250) — compile-verified, matrix operand order checked by reading, not run on hardware
//
#include <hip/hip_runtime.h>

// ---------------- problem constants ----------------
constexpr int NB   = 32;     // batch
constexpr int CIN  = 1024;   // input channels (== W in grid_sample view)
constexpr int TT   = 512;    // sequence length (== H)
constexpr int COUT = 1024;
constexpr int KW   = 3;      // kernel taps
constexpr int M    = NB * TT;      // 16384 GEMM rows
constexpr int KD   = KW * CIN;     // 3072 GEMM reduction dim

// ---------------- WMMA types ----------------
typedef __attribute__((ext_vector_type(16))) __bf16 v16bf;
typedef __attribute__((ext_vector_type(8)))  float  v8f;

union FragU {
    v16bf v;
    uint4 q[2];
};

__device__ __forceinline__ unsigned short f2bf(float f) {
    unsigned int u = __float_as_uint(f);
    unsigned int r = (u + 0x7FFFu + ((u >> 16) & 1u)) >> 16;  // RNE
    return (unsigned short)r;
}

// ---------------- kernel 1: transpose inputs (B,C,T) -> inT (B,T,C) ----------------
__global__ void transpose_kernel(const float* __restrict__ in, float* __restrict__ inT) {
    __shared__ float tile[32][33];
    int b  = blockIdx.z;
    int t0 = blockIdx.x * 32;
    int c0 = blockIdx.y * 32;
    int lx = threadIdx.x;        // 0..31
    int ly = threadIdx.y;        // 0..7
    const float* src = in + (size_t)b * CIN * TT;
    float*       dst = inT + (size_t)b * TT * CIN;
#pragma unroll
    for (int i = 0; i < 32; i += 8)
        tile[ly + i][lx] = src[(size_t)(c0 + ly + i) * TT + (t0 + lx)];
    __syncthreads();
#pragma unroll
    for (int i = 0; i < 32; i += 8)
        dst[(size_t)(t0 + ly + i) * CIN + (c0 + lx)] = tile[lx][ly + i];
}

// ---------------- kernel 2: offset conv + grid math -> y0c,y1c,ty ----------------
__global__ void offsets_kernel(const float* __restrict__ in,
                               const float* __restrict__ w_off,
                               const float* __restrict__ b_off,
                               int* __restrict__ y0c, int* __restrict__ y1c,
                               float* __restrict__ tyv) {
    int g = blockIdx.x * blockDim.x + threadIdx.x;  // 0..M-1
    int b = g >> 9;
    int t = g & (TT - 1);
    const float* xb = in + (size_t)b * CIN * TT;
    float acc0 = 0.f, acc1 = 0.f, acc2 = 0.f;
    for (int c = 0; c < CIN; ++c) {
        float xm = (t > 0)      ? xb[(size_t)c * TT + t - 1] : 0.f;
        float x0 = xb[(size_t)c * TT + t];
        float xp = (t < TT - 1) ? xb[(size_t)c * TT + t + 1] : 0.f;
        const float* wp = w_off + (size_t)c * 3;   // w_off[k][c][j], k-stride = CIN*3
        acc0 += xm * wp[0]             + x0 * wp[1]             + xp * wp[2];
        acc1 += xm * wp[CIN * 3 + 0]   + x0 * wp[CIN * 3 + 1]   + xp * wp[CIN * 3 + 2];
        acc2 += xm * wp[2 * CIN * 3]   + x0 * wp[2 * CIN * 3+1] + xp * wp[2 * CIN * 3 + 2];
    }
    float pred[3] = {acc0 + b_off[0], acc1 + b_off[1], acc2 + b_off[2]};
    float base = ((float)t / (float)(TT - 1)) * 2.f - 1.f;
#pragma unroll
    for (int k = 0; k < 3; ++k) {
        float gg = base + (float)(k - 1) * (2.f / (float)(TT - 1)) + pred[k];
        gg = fminf(1.f, fmaxf(-1.f, gg));
        float py = ((gg + 1.f) * (float)TT - 1.f) * 0.5f;
        float y0 = floorf(py);
        float ty = py - y0;
        int y0i = (int)y0;
        int a  = min(max(y0i, 0), TT - 1);
        int bb = min(max(y0i + 1, 0), TT - 1);
        int idx = g * 3 + k;
        y0c[idx] = a;
        y1c[idx] = bb;
        tyv[idx] = ty;
    }
}

// -------- kernel 3: fold channel-mix into weights; bf16 B-matrix [COUT][KD] --------
__global__ void wtrans_kernel(const float* __restrict__ w_conv,
                              unsigned short* __restrict__ wB) {
    int tid = blockIdx.x * blockDim.x + threadIdx.x;  // o*KD + k*CIN + c
    if (tid >= COUT * KD) return;
    int o   = tid / KD;
    int rem = tid - o * KD;
    int k   = rem / CIN;
    int c   = rem - k * CIN;
    float acc = 0.f;
#pragma unroll
    for (int dw = -1; dw <= 1; ++dw) {
        int w = c + dw;
        if (w < 0 || w >= CIN) continue;
        float px  = (float)w * ((float)CIN / (float)(CIN - 1)) - 0.5f;
        float x0  = floorf(px);
        float tx  = px - x0;
        int x0i   = (int)x0;
        int x0cl  = min(max(x0i, 0), CIN - 1);
        int x1cl  = min(max(x0i + 1, 0), CIN - 1);
        float wv  = w_conv[(size_t)o * CIN * 3 + (size_t)w * 3 + k];
        if (x0cl == c) acc += (1.f - tx) * wv;
        if (x1cl == c) acc += tx * wv;
    }
    wB[tid] = f2bf(acc);
}

// -------- kernel 4: y-bilinear sampling -> bf16 A-matrix [M][KD] --------
__global__ void sample_kernel(const float* __restrict__ inT,
                              const int* __restrict__ y0c,
                              const int* __restrict__ y1c,
                              const float* __restrict__ tyv,
                              unsigned short* __restrict__ sampA) {
    int row = blockIdx.x;   // 0..M-1
    int k   = blockIdx.y;   // 0..2
    int b   = row >> 9;
    int idx = row * 3 + k;
    int y0 = y0c[idx];
    int y1 = y1c[idx];
    float ty = tyv[idx];
    const float* r0 = inT + ((size_t)b * TT + y0) * CIN;
    const float* r1 = inT + ((size_t)b * TT + y1) * CIN;
    unsigned short* dst = sampA + (size_t)row * KD + (size_t)k * CIN;
    for (int c = threadIdx.x; c < CIN; c += blockDim.x) {
        float v = (1.f - ty) * r0[c] + ty * r1[c];
        dst[c] = f2bf(v);
    }
}

// ------- kernel 5: bf16 WMMA GEMM + bias (async global->LDS double buffer) -------
constexpr int BM  = 128;
constexpr int BN  = 128;
constexpr int BK  = 64;
constexpr int LDA = 80;  // 64 + 16 pad (row stride 160B keeps 16B alignment)

__device__ __forceinline__ void async_ld128(unsigned lds_byte, unsigned goff_byte,
                                            const void* base) {
    asm volatile("global_load_async_to_lds_b128 %0, %1, %2"
                 :
                 : "v"(lds_byte), "v"(goff_byte), "s"(base)
                 : "memory");
}
__device__ __forceinline__ void wait_async0() {
    asm volatile("s_wait_asynccnt 0x0" ::: "memory");
}

__global__ void __launch_bounds__(256)
gemm_kernel(const unsigned short* __restrict__ A,   // [M][KD] bf16
            const unsigned short* __restrict__ Bw,  // [COUT][KD] bf16 (K-major)
            const float* __restrict__ bias,
            float* __restrict__ out) {              // [M][COUT] f32
    __shared__ __attribute__((aligned(16))) unsigned short sA[2][BM * LDA];
    __shared__ __attribute__((aligned(16))) unsigned short sB[2][BN * LDA];

    int m0 = blockIdx.x * BM;
    int n0 = blockIdx.y * BN;
    int tid  = threadIdx.x;
    int wave = tid >> 5;
    int lane = tid & 31;
    int mw   = (wave & 3) * 32;   // wave M offset in tile
    int nw   = (wave >> 2) * 64;  // wave N offset in tile
    int lr   = lane & 15;
    int half = lane >> 4;

    constexpr unsigned BUFA = (unsigned)(BM * LDA * 2);  // bytes per A buffer
    constexpr unsigned BUFB = (unsigned)(BN * LDA * 2);

    v8f acc[2][4];
#pragma unroll
    for (int i = 0; i < 2; ++i)
#pragma unroll
        for (int j = 0; j < 4; ++j)
#pragma unroll
            for (int e = 0; e < 8; ++e) acc[i][j][e] = 0.f;

    // per-thread staging: 4 x 16B chunks of each 128x64 tile (byte offsets)
    unsigned gA[4], gB[4], lA[4], lB[4];
#pragma unroll
    for (int i = 0; i < 4; ++i) {
        int ch = tid + i * 256;         // 0..1023
        int r  = ch >> 3;
        int c8 = (ch & 7) * 8;
        gA[i] = (unsigned)(((m0 + r) * KD + c8) * 2);
        gB[i] = (unsigned)(((n0 + r) * KD + c8) * 2);
        // low 32 bits of the generic shared pointer == LDS byte address
        lA[i] = (unsigned)(size_t)&sA[0][r * LDA + c8];
        lB[i] = (unsigned)(size_t)&sB[0][r * LDA + c8];
    }

    // prologue: async-copy K-step 0 into buffer 0
#pragma unroll
    for (int i = 0; i < 4; ++i) {
        async_ld128(lA[i], gA[i], A);
        async_ld128(lB[i], gB[i], Bw);
    }
    wait_async0();
    __syncthreads();

    int cur = 0;
    for (int kk = BK; kk <= KD; kk += BK) {
        bool more = (kk < KD);          // uniform across block
        if (more) {
            // async-copy next tile into the other buffer (overlaps WMMA below)
            unsigned kb  = (unsigned)kk * 2u;
            unsigned nxt = (unsigned)(cur ^ 1);
#pragma unroll
            for (int i = 0; i < 4; ++i) {
                async_ld128(lA[i] + nxt * BUFA, gA[i] + kb, A);
                async_ld128(lB[i] + nxt * BUFB, gB[i] + kb, Bw);
            }
        }

        // compute on buffer `cur`
#pragma unroll
        for (int ks = 0; ks < 2; ++ks) {
            int colb = ks * 32 + half * 8;  // K chunk base for this lane-half
            FragU af[2], bfm[4];
#pragma unroll
            for (int mt = 0; mt < 2; ++mt) {
                const unsigned short* p = &sA[cur][(mw + mt * 16 + lr) * LDA];
                af[mt].q[0] = *(const uint4*)(p + colb);
                af[mt].q[1] = *(const uint4*)(p + colb + 16);
            }
#pragma unroll
            for (int nt = 0; nt < 4; ++nt) {
                const unsigned short* p = &sB[cur][(nw + nt * 16 + lr) * LDA];
                bfm[nt].q[0] = *(const uint4*)(p + colb);
                bfm[nt].q[1] = *(const uint4*)(p + colb + 16);
            }
#pragma unroll
            for (int mt = 0; mt < 2; ++mt)
#pragma unroll
                for (int nt = 0; nt < 4; ++nt)
                    acc[mt][nt] = __builtin_amdgcn_wmma_f32_16x16x32_bf16(
                        false, af[mt].v, false, bfm[nt].v,
                        (short)0, acc[mt][nt], false, false);
        }

        if (more) {
            wait_async0();              // own async copies landed in LDS
            __syncthreads();            // all waves' copies visible
            cur ^= 1;
        }
    }

    // epilogue: C layout — VGPR r holds M = r + 8*half, N = lane%16
#pragma unroll
    for (int nt = 0; nt < 4; ++nt) {
        int n = n0 + nw + nt * 16 + lr;
        float bv = bias[n];
#pragma unroll
        for (int mt = 0; mt < 2; ++mt) {
#pragma unroll
            for (int r = 0; r < 8; ++r) {
                int m = m0 + mw + mt * 16 + r + 8 * half;
                out[(size_t)m * COUT + n] = acc[mt][nt][r] + bv;
            }
        }
    }
}

// ---------------- launch ----------------
extern "C" void kernel_launch(void* const* d_in, const int* in_sizes, int n_in,
                              void* d_out, int out_size, void* d_ws, size_t ws_size,
                              hipStream_t stream) {
    const float* in     = (const float*)d_in[0];
    const float* w_off  = (const float*)d_in[1];
    const float* b_off  = (const float*)d_in[2];
    const float* w_conv = (const float*)d_in[3];
    const float* b_conv = (const float*)d_in[4];
    float* out = (float*)d_out;

    constexpr size_t SZ_A = (size_t)M * KD * sizeof(unsigned short);    // 100663296
    constexpr size_t SZ_B = (size_t)COUT * KD * sizeof(unsigned short); //   6291456
    constexpr size_t SZ_T = (size_t)NB * TT * CIN * sizeof(float);      //  67108864
    constexpr size_t SZ_I = (size_t)M * 3 * sizeof(int);                //    196608

    char* ws = (char*)d_ws;
    unsigned short* sampA = (unsigned short*)(ws);
    unsigned short* wB    = (unsigned short*)(ws + SZ_A);
    float*          inT   = (float*)(ws + SZ_A + SZ_B);
    int*            y0c   = (int*)(ws + SZ_A + SZ_B + SZ_T);
    int*            y1c   = (int*)(ws + SZ_A + SZ_B + SZ_T + SZ_I);
    float*          tyv   = (float*)(ws + SZ_A + SZ_B + SZ_T + 2 * SZ_I);

    // 1) transpose inputs -> inT
    transpose_kernel<<<dim3(TT / 32, CIN / 32, NB), dim3(32, 8), 0, stream>>>(in, inT);
    // 2) offset conv + grid math
    offsets_kernel<<<dim3(M / 256), dim3(256), 0, stream>>>(in, w_off, b_off, y0c, y1c, tyv);
    // 3) weight fold -> bf16 B matrix
    wtrans_kernel<<<dim3((COUT * KD) / 256), dim3(256), 0, stream>>>(w_conv, wB);
    // 4) bilinear sampling -> bf16 A matrix
    sample_kernel<<<dim3(M, 3), dim3(256), 0, stream>>>(inT, y0c, y1c, tyv, sampA);
    // 5) WMMA GEMM + bias
    gemm_kernel<<<dim3(M / BM, COUT / BN), dim3(256), 0, stream>>>(sampA, wB, b_conv, out);
}
